// GNN_46007689675004
// MI455X (gfx1250) — matrix operand order repaired and checked by reading, not compiled, not run on hardware
//
#include <hip/hip_runtime.h>
#include <hip/hip_bf16.h>

#define BATCH   8
#define NPRED   25500
#define NC      80
#define CP      85
#define MAXDET  300
#define FDIM    1920
#define HDIM    256
#define OUTC    260

typedef _Float16 f16;
typedef __attribute__((ext_vector_type(16))) _Float16 v16h;
typedef __attribute__((ext_vector_type(8)))  _Float16 v8h;
typedef __attribute__((ext_vector_type(8)))  float    v8f;

// ---------------- Kernel 1: conf = obj * max(cls), argmax, threshold ----------------
__global__ void score_kernel(const float* __restrict__ preds,
                             float* __restrict__ conf, int* __restrict__ cls) {
    int t = blockIdx.x * blockDim.x + threadIdx.x;
    if (t >= BATCH * NPRED) return;
    const float* p = preds + (size_t)t * CP;
    float obj = p[4];
    float best = -1.0f; int bi = 0;
    for (int c = 0; c < NC; ++c) {
        float s = p[5 + c];
        if (s > best) { best = s; bi = c; }
    }
    float cf = best * obj;
    conf[t] = (cf > 0.1f) ? cf : 0.0f;
    cls[t] = bi;
}

// ---------------- Kernel 2: per-batch top-300 via iterative LDS max-reduce ----------
__global__ void __launch_bounds__(1024)
topk_kernel(float* __restrict__ conf, float* __restrict__ topVal, int* __restrict__ topIdx) {
    __shared__ float sv[1024];
    __shared__ int   si[1024];
    int b = blockIdx.x;
    float* cf = conf + (size_t)b * NPRED;
    int tid = threadIdx.x;
    for (int r = 0; r < MAXDET; ++r) {
        float bv = -2.0f; int bi = 0;
        for (int i = tid; i < NPRED; i += 1024) {
            float v = cf[i];
            if (v > bv || (v == bv && i < bi)) { bv = v; bi = i; }
        }
        sv[tid] = bv; si[tid] = bi;
        __syncthreads();
        for (int s = 512; s > 0; s >>= 1) {
            if (tid < s) {
                float ov = sv[tid + s]; int oi = si[tid + s];
                if (ov > sv[tid] || (ov == sv[tid] && oi < si[tid])) { sv[tid] = ov; si[tid] = oi; }
            }
            __syncthreads();
        }
        if (tid == 0) {
            topVal[b * MAXDET + r] = sv[0];
            topIdx[b * MAXDET + r] = si[0];
            cf[si[0]] = -1.0f;       // remove winner for next round
            __threadfence_block();
        }
        __syncthreads();
    }
}

// ---------------- Kernel 3: xywh->xyxy, class offset, greedy NMS in LDS -------------
__global__ void __launch_bounds__(320)
nms_kernel(const float* __restrict__ preds, const int* __restrict__ cls,
           const float* __restrict__ topVal, const int* __restrict__ topIdx,
           float* __restrict__ boxesOut, int* __restrict__ keepOut) {
    __shared__ float bx[MAXDET][4];
    __shared__ int   kp[MAXDET];
    int b = blockIdx.x, tid = threadIdx.x;
    for (int j = tid; j < MAXDET; j += blockDim.x) {
        int i = topIdx[b * MAXDET + j];
        const float* p = preds + ((size_t)b * NPRED + i) * CP;
        float cx = p[0], cy = p[1], w = p[2], h = p[3];
        float x1 = cx - w * 0.5f, y1 = cy - h * 0.5f;
        float x2 = cx + w * 0.5f, y2 = cy + h * 0.5f;
        float* bo = boxesOut + (size_t)(b * MAXDET + j) * 4;
        bo[0] = x1; bo[1] = y1; bo[2] = x2; bo[3] = y2;
        float offc = (float)cls[b * NPRED + i] * 4096.0f;
        bx[j][0] = x1 + offc; bx[j][1] = y1 + offc;
        bx[j][2] = x2 + offc; bx[j][3] = y2 + offc;
        kp[j] = (topVal[b * MAXDET + j] > 0.1f) ? 1 : 0;
    }
    __syncthreads();
    for (int i = 0; i < MAXDET; ++i) {
        if (kp[i]) {
            int j = tid;
            if (j < MAXDET && j > i && kp[j]) {
                float ax1 = bx[i][0], ay1 = bx[i][1], ax2 = bx[i][2], ay2 = bx[i][3];
                float cx1 = bx[j][0], cy1 = bx[j][1], cx2 = bx[j][2], cy2 = bx[j][3];
                float areaA = (ax2 - ax1) * (ay2 - ay1);
                float areaB = (cx2 - cx1) * (cy2 - cy1);
                float iw = fminf(ax2, cx2) - fmaxf(ax1, cx1); iw = iw > 0.f ? iw : 0.f;
                float ih = fminf(ay2, cy2) - fmaxf(ay1, cy1); ih = ih > 0.f ? ih : 0.f;
                float inter = iw * ih;
                float iou = inter / (areaA + areaB - inter + 1e-7f);
                if (iou > 0.6f) kp[j] = 0;
            }
        }
        __syncthreads();
    }
    for (int j = tid; j < MAXDET; j += blockDim.x) keepOut[b * MAXDET + j] = kp[j];
}

// ---------------- Kernel 4: 1x1 ROI align over 4 levels -> f16 activations ---------
__global__ void __launch_bounds__(256)
roi_kernel(const float* __restrict__ f1, const float* __restrict__ f2,
           const float* __restrict__ f3, const float* __restrict__ f4,
           const float* __restrict__ boxes, f16* __restrict__ X) {
    int blk = blockIdx.x;                // b * MAXDET + j
    int b = blk / MAXDET;
    const float* bo = boxes + (size_t)blk * 4;
    float x1 = bo[0], y1 = bo[1], x2 = bo[2], y2 = bo[3];
    for (int c = threadIdx.x; c < FDIM; c += blockDim.x) {
        const float* feat; int H, W; float scale; int lc;
        if (c < 128)      { lc = c;       feat = f1 + (size_t)b * 128 * 80 * 80;  H = 80; W = 80; scale = 0.125f;    }
        else if (c < 384) { lc = c - 128; feat = f2 + (size_t)b * 256 * 40 * 40;  H = 40; W = 40; scale = 0.0625f;   }
        else if (c < 896) { lc = c - 384; feat = f3 + (size_t)b * 512 * 20 * 20;  H = 20; W = 20; scale = 0.03125f;  }
        else              { lc = c - 896; feat = f4 + (size_t)b * 1024 * 10 * 10; H = 10; W = 10; scale = 0.015625f; }
        float bx1 = x1 * scale, by1 = y1 * scale, bx2 = x2 * scale, by2 = y2 * scale;
        float w = fmaxf(bx2 - bx1, 1.0f), h = fmaxf(by2 - by1, 1.0f);
        float xs0 = bx1 + 0.5f * (w * 0.5f), xs1 = bx1 + 1.5f * (w * 0.5f);
        float ys0 = by1 + 0.5f * (h * 0.5f), ys1 = by1 + 1.5f * (h * 0.5f);
        const float* ch = feat + (size_t)lc * H * W;
        float yy[4] = { ys0, ys0, ys1, ys1 };
        float xx[4] = { xs0, xs1, xs0, xs1 };
        float acc = 0.0f;
        for (int s = 0; s < 4; ++s) {
            float y = fminf(fmaxf(yy[s], 0.0f), (float)(H - 1));
            float x = fminf(fmaxf(xx[s], 0.0f), (float)(W - 1));
            int y0 = (int)floorf(y), x0 = (int)floorf(x);
            int y1i = min(y0 + 1, H - 1), x1i = min(x0 + 1, W - 1);
            float ly = y - (float)y0, lx = x - (float)x0;
            acc += ch[y0 * W + x0] * ((1.f - ly) * (1.f - lx))
                 + ch[y0 * W + x1i] * ((1.f - ly) * lx)
                 + ch[y1i * W + x0] * (ly * (1.f - lx))
                 + ch[y1i * W + x1i] * (ly * lx);
        }
        X[(size_t)blk * FDIM + c] = (f16)(acc * 0.25f);
    }
}

// ---------------- Kernel 5: weight transpose + f32 -> f16 convert -------------------
__global__ void wt_kernel(const float* __restrict__ Wsrc, f16* __restrict__ Wt, int K, int N) {
    int t = blockIdx.x * blockDim.x + threadIdx.x;
    if (t >= K * N) return;
    int k = t / N, n = t % N;
    Wt[(size_t)n * K + k] = (f16)Wsrc[t];
}

// ---------------- Kernel 6/7: WMMA GEMM (16x16 tile per wave, K-loop of 32) ---------
// A layout (f16 16x32): lane<16 -> M=lane, K 0..15; lane>=16 -> M=lane-16, K 16..31
// B layout (f16 32x16): lane<16 -> N=lane, K 0..15; lane>=16 -> N=lane-16, K 16..31
// C layout (f32 16x16): VGPR r -> row r + 8*(lane>=16), col lane&15
template <int KDIM>
__global__ void __launch_bounds__(256)
gemm_wmma(const f16* __restrict__ Xm, const f16* __restrict__ Wt,
          const float* __restrict__ bias, float* __restrict__ outF, f16* __restrict__ outH) {
    int wave = threadIdx.x >> 5;
    int lane = threadIdx.x & 31;
    int m0 = blockIdx.x * 16;
    int n0 = blockIdx.y * 128 + wave * 16;
    int loc = lane & 15;
    int hi  = lane >> 4;                                  // 0 or 1
    const f16* arow = Xm + (size_t)(m0 + loc) * KDIM + hi * 16;
    const f16* brow = Wt + (size_t)(n0 + loc) * KDIM + hi * 16;
    v8f c = {};
    for (int k0 = 0; k0 < KDIM; k0 += 32) {
        if (k0 + 64 < KDIM) {                             // gfx1250 global_prefetch_b8
            __builtin_prefetch(arow + k0 + 64, 0, 1);
            __builtin_prefetch(brow + k0 + 64, 0, 1);
        }
        union { v16h v; v8h h[2]; } a, bm;
        a.h[0]  = *(const v8h*)(arow + k0);               // K base .. base+7
        a.h[1]  = *(const v8h*)(arow + k0 + 8);           // K base+8 .. base+15
        bm.h[0] = *(const v8h*)(brow + k0);               // K kb .. kb+7
        bm.h[1] = *(const v8h*)(brow + k0 + 8);           // K kb+8 .. kb+15
        c = __builtin_amdgcn_wmma_f32_16x16x32_f16(false, a.v, false, bm.v,
                                                   (short)0, c, false, false);
    }
    float bv = bias[n0 + loc];
    int rbase = m0 + hi * 8;
    for (int r = 0; r < 8; ++r) {
        float v = c[r] + bv;
        v = (v > 0.0f) ? v : 0.01f * v;                   // leaky relu
        size_t o = (size_t)(rbase + r) * HDIM + (n0 + loc);
        if (outF) outF[o] = v;
        if (outH) outH[o] = (f16)v;
    }
}

// ---------------- Kernel 8: rescale boxes, concat, keep-mask ------------------------
__global__ void finalize_kernel(const float* __restrict__ boxes, const int* __restrict__ keep,
                                const float* __restrict__ H2, const float* __restrict__ orig_hw,
                                float* __restrict__ out) {
    int t = blockIdx.x * blockDim.x + threadIdx.x;
    if (t >= BATCH * MAXDET * OUTC) return;
    int c  = t % OUTC;
    int bj = t / OUTC;
    int b  = bj / MAXDET;
    float kf = keep[bj] ? 1.0f : 0.0f;
    float v;
    if (c < 4) {
        float h0 = orig_hw[b * 2 + 0], w0 = orig_hw[b * 2 + 1];
        float gain  = fminf(640.0f / h0, 640.0f / w0);
        float pad_x = (640.0f - w0 * gain) * 0.5f;
        float pad_y = (640.0f - h0 * gain) * 0.5f;
        float coord = boxes[(size_t)bj * 4 + c];
        float pad = (c == 0 || c == 2) ? pad_x : pad_y;
        float lim = (c == 0 || c == 2) ? w0 : h0;
        v = fminf(fmaxf((coord - pad) / gain, 0.0f), lim) / lim;
    } else {
        v = H2[(size_t)bj * HDIM + (c - 4)];
    }
    out[t] = v * kf;
}

extern "C" void kernel_launch(void* const* d_in, const int* in_sizes, int n_in,
                              void* d_out, int out_size, void* d_ws, size_t ws_size,
                              hipStream_t stream) {
    const float* preds   = (const float*)d_in[0];
    const float* feat1   = (const float*)d_in[1];
    const float* feat2   = (const float*)d_in[2];
    const float* feat3   = (const float*)d_in[3];
    const float* feat4   = (const float*)d_in[4];
    const float* orig_hw = (const float*)d_in[5];
    const float* W1      = (const float*)d_in[6];
    const float* b1      = (const float*)d_in[7];
    const float* W2      = (const float*)d_in[8];
    const float* b2      = (const float*)d_in[9];
    float* out = (float*)d_out;

    char* ws = (char*)d_ws;
    size_t off = 0;
    auto carve = [&](size_t bytes) -> void* {
        void* p = ws + off;
        off = (off + bytes + 255) & ~(size_t)255;
        return p;
    };
    float* conf   = (float*)carve((size_t)BATCH * NPRED * sizeof(float));
    int*   cls    = (int*)  carve((size_t)BATCH * NPRED * sizeof(int));
    int*   topIdx = (int*)  carve((size_t)BATCH * MAXDET * sizeof(int));
    float* topVal = (float*)carve((size_t)BATCH * MAXDET * sizeof(float));
    float* boxes  = (float*)carve((size_t)BATCH * MAXDET * 4 * sizeof(float));
    int*   keep   = (int*)  carve((size_t)BATCH * MAXDET * sizeof(int));
    f16*   X      = (f16*)  carve((size_t)BATCH * MAXDET * FDIM * sizeof(f16));
    f16*   Wt1    = (f16*)  carve((size_t)HDIM * FDIM * sizeof(f16));
    f16*   Wt2    = (f16*)  carve((size_t)HDIM * HDIM * sizeof(f16));
    f16*   H1     = (f16*)  carve((size_t)BATCH * MAXDET * HDIM * sizeof(f16));
    float* H2     = (float*)carve((size_t)BATCH * MAXDET * HDIM * sizeof(float));

    score_kernel<<<(BATCH * NPRED + 255) / 256, 256, 0, stream>>>(preds, conf, cls);
    topk_kernel<<<BATCH, 1024, 0, stream>>>(conf, topVal, topIdx);
    nms_kernel<<<BATCH, 320, 0, stream>>>(preds, cls, topVal, topIdx, boxes, keep);
    roi_kernel<<<BATCH * MAXDET, 256, 0, stream>>>(feat1, feat2, feat3, feat4, boxes, X);
    wt_kernel<<<(FDIM * HDIM + 255) / 256, 256, 0, stream>>>(W1, Wt1, FDIM, HDIM);
    wt_kernel<<<(HDIM * HDIM + 255) / 256, 256, 0, stream>>>(W2, Wt2, HDIM, HDIM);

    dim3 gGemm((BATCH * MAXDET) / 16, HDIM / 128);       // 150 x 2 blocks, 8 waves each
    gemm_wmma<FDIM><<<gGemm, 256, 0, stream>>>(X, Wt1, b1, nullptr, H1);
    gemm_wmma<HDIM><<<gGemm, 256, 0, stream>>>(H1, Wt2, b2, H2, nullptr);

    finalize_kernel<<<(BATCH * MAXDET * OUTC + 255) / 256, 256, 0, stream>>>(
        boxes, keep, H2, orig_hw, out);
}
